// Encoder_38087769981494
// MI455X (gfx1250) — matrix operand order repaired and checked by reading, not compiled, not run on hardware
//
#include <hip/hip_runtime.h>
#include <hip/hip_bf16.h>

// ---------------------------------------------------------------------------
// Encoder: concat -> 4x [dilated conv1d(k=3, VALID) -> BN(train) -> LeakyReLU]
// Conv as WMMA GEMM: M = time, N = c_out, K = (tap, c_in).
// Intermediates (B, T, C) bf16 (channel-contiguous); weights packed
// (C_out, 3*Cin) bf16 -> WMMA fragments load with no transposes.
// Input tiles staged to LDS by the Tensor Data Mover (double-buffered),
// compute = v_wmma_f32_16x16x32_bf16, BN stats fused in the epilogue.
// ---------------------------------------------------------------------------

typedef __attribute__((ext_vector_type(16))) __bf16       v16bf;
typedef __attribute__((ext_vector_type(8)))  float        v8f;
typedef __attribute__((ext_vector_type(4)))  __bf16       v4bf;
typedef __attribute__((ext_vector_type(4)))  unsigned int v4u;
typedef __attribute__((ext_vector_type(8)))  int          v8i;
typedef __attribute__((ext_vector_type(4)))  int          v4i;

static constexpr int kB    = 32;
static constexpr int kT    = 4096;
static constexpr int kCout = 256;

// --------------------------- TDM 2D tile loader ----------------------------
// Load a [tile_d1 rows x tile_d0 cols] tile of 2-byte elements from a
// row-major (tensor_d1 x tensor_d0) tensor into LDS (packed row-major).
// Rows beyond tensor_d1 / cols beyond tensor_d0 read as zero (OOB rule).
__device__ __forceinline__ void tdm_load_2d_bf16(unsigned lds_addr,
                                                 const void* gaddr,
                                                 int tensor_d0, int tensor_d1,
                                                 int tile_d0, int tile_d1,
                                                 int stride_d0)
{
    const unsigned long long ga = (unsigned long long)(uintptr_t)gaddr;
    v4u g0;
    g0[0] = 1u;                                        // count=1, user mode
    g0[1] = lds_addr;                                  // LDS byte address
    g0[2] = (unsigned)(ga & 0xFFFFFFFFu);              // global_addr[31:0]
    g0[3] = ((unsigned)(ga >> 32) & 0x01FFFFFFu)       // global_addr[56:32]
            | 0x80000000u;                             // type = 2 ("image")
    v8i g1;
    g1[0] = 0x00010000;                                // data_size = 2 bytes
    g1[1] = (tensor_d0 & 0xFFFF) << 16;                // tensor_dim0[15:0]
    g1[2] = ((tensor_d0 >> 16) & 0xFFFF)               // tensor_dim0[31:16]
            | ((tensor_d1 & 0xFFFF) << 16);            // tensor_dim1[15:0]
    g1[3] = ((tensor_d1 >> 16) & 0xFFFF)               // tensor_dim1[31:16]
            | ((tile_d0 & 0xFFFF) << 16);              // tile_dim0
    g1[4] = tile_d1 & 0xFFFF;                          // tile_dim1 (tile_dim2=0)
    g1[5] = stride_d0;                                 // tensor_dim0_stride lo32
    g1[6] = 0;                                         // stride hi16 | dim1_stride lo16
    g1[7] = 0;                                         // dim1_stride hi32
    const v4i gz4 = {0, 0, 0, 0};
    const v8i gz8 = {0, 0, 0, 0, 0, 0, 0, 0};
    // 6-arg form on this toolchain: (g0, g1, g2, g3, g4, cpol)
    __builtin_amdgcn_tensor_load_to_lds(g0, g1, gz4, gz4, gz8, 0);
}

// -------------------------------- concat -----------------------------------
// X:(B,48,T) f32, Y:(B,16,T) f32 -> h:(B,T,64) bf16 (transpose via LDS)
__global__ __launch_bounds__(256)
void enc_concat_bf16(const float* __restrict__ X, const float* __restrict__ Y,
                     __bf16* __restrict__ h)
{
    __shared__ float tile[32][33];
    const int b  = blockIdx.z;
    const int t0 = blockIdx.x << 5;
    const int c0 = blockIdx.y << 5;
    const int tx = threadIdx.x & 31;
    const int ty = threadIdx.x >> 5;
#pragma unroll
    for (int j = 0; j < 4; ++j) {
        const int c = c0 + ty + (j << 3);
        const int t = t0 + tx;
        float v;
        if (c < 48) v = X[((size_t)b * 48 + c) * kT + t];
        else        v = Y[((size_t)b * 16 + (c - 48)) * kT + t];
        tile[ty + (j << 3)][tx] = v;
    }
    __syncthreads();
#pragma unroll
    for (int j = 0; j < 4; ++j) {
        const int t = t0 + ty + (j << 3);
        const int c = c0 + tx;
        h[((size_t)b * kT + t) * 64 + c] = (__bf16)tile[tx][ty + (j << 3)];
    }
}

// ----------------------------- weight packing ------------------------------
// W:(Cout,Cin,3) f32 -> wpk:(Cout, 3*Cin) bf16, K index = k*Cin + ci
__global__ void enc_pack_w(const float* __restrict__ W, __bf16* __restrict__ wpk,
                           int Cin)
{
    const int total = kCout * Cin * 3;
    const int i = blockIdx.x * blockDim.x + threadIdx.x;
    if (i >= total) return;
    const int k  = i % 3;
    const int ci = (i / 3) % Cin;
    const int co = i / (3 * Cin);
    wpk[(size_t)co * (3 * Cin) + k * Cin + ci] = (__bf16)W[i];
}

__global__ void enc_zero_f32(float* __restrict__ p, int n)
{
    const int i = blockIdx.x * blockDim.x + threadIdx.x;
    if (i < n) p[i] = 0.f;
}

// --------------------------- conv (WMMA) + stats ---------------------------
// Block = 8 waves: 2 (time) x 4 (c_out). Block tile 128t x 128co.
// Wave tile 64t x 32co -> 8 accumulators, 8 WMMA per K-chunk.
// A tile (128 rows x 32 K bf16) staged to LDS by TDM, double-buffered.
template<int CIN, int DIL, int TIN, int TOUT>
__global__ __launch_bounds__(256)
void enc_conv_wmma_t(const __bf16* __restrict__ in,
                     const __bf16* __restrict__ wpk,
                     float* __restrict__ raw,
                     float* __restrict__ stats)
{
    constexpr int KPK = 3 * CIN;          // packed K extent
    constexpr int CPK = CIN / 32;         // K-chunks per tap
    constexpr int KCH = 3 * CPK;          // total K-chunks
    constexpr int TILE_ELems = 128 * 32;  // LDS tile elements per buffer

    const int lane = threadIdx.x & 31;
    const int wave = threadIdx.x >> 5;
    const int tw   = wave >> 2;                   // 0..1 (time sub-tile)
    const int cw   = wave & 3;                    // 0..3 (cout sub-tile)
    const int b    = blockIdx.z;
    const int t0   = blockIdx.x << 7;             // 128 time rows / block
    const int co0  = (blockIdx.y << 7) + (cw << 5);

    __shared__ __align__(64) __bf16 smA[2 * TILE_ELems];

    const __bf16* __restrict__ inB = in + (size_t)b * TIN * CIN;
    const unsigned ldsBase = (unsigned)(uintptr_t)(&smA[0]);

    const v8f vz = {0.f, 0.f, 0.f, 0.f, 0.f, 0.f, 0.f, 0.f};
    v8f acc[4][2];
#pragma unroll
    for (int mt = 0; mt < 4; ++mt) { acc[mt][0] = vz; acc[mt][1] = vz; }

    const int halfB = (lane >> 4) << 4;           // B frag: +16 K for hi lanes
    const int halfA = (lane >> 4) << 3;           // unused; A frag also uses halfB (+16 K)
    const int rowA0 = (tw << 6) + (lane & 15);    // wave's base row in tile

    // TDM issue for K-chunk kc into LDS buffer `buf`.
    auto issue = [&](int kc, int buf) {
        const int k    = kc / CPK;                        // tap 0..2
        const int ci0  = (kc - k * CPK) << 5;             // c_in base
        const int trow = t0 + k * DIL;                    // first time row
        int remain = TIN - trow; if (remain < 0) remain = 0;
        tdm_load_2d_bf16(ldsBase + (unsigned)buf * (TILE_ELems * 2),
                         inB + (size_t)trow * CIN + ci0,
                         /*tensor_d0=*/CIN, /*tensor_d1=*/remain,
                         /*tile_d0=*/32, /*tile_d1=*/128,
                         /*stride_d0=*/CIN);
    };

    if (wave == 0) issue(0, 0);

#pragma unroll 1
    for (int kc = 0; kc < KCH; ++kc) {
        if (wave == 0) __builtin_amdgcn_s_wait_tensorcnt(0);
        __syncthreads();                           // tile kc visible to all
        if (wave == 0 && kc + 1 < KCH) issue(kc + 1, (kc + 1) & 1);

        const __bf16* sA = smA + (kc & 1) * TILE_ELems;

        // Two B fragments (32K x 16co each): lane holds column co+(lane&15),
        // 16 contiguous K values starting at kc*32 + (lane>=16)*16.
        const __bf16* wr0 =
            wpk + (size_t)(co0 + (lane & 15)) * KPK + (kc << 5) + halfB;
        const v16bf bf0 = *(const v16bf*)wr0;
        const v16bf bf1 = *(const v16bf*)(wr0 + (size_t)16 * KPK);
        if (kc + 1 < KCH) __builtin_prefetch(wr0 + 32, 0, 0);

#pragma unroll
        for (int mt = 0; mt < 4; ++mt) {
            // A fragment row M = rowA0 + mt*16; lanes 0-15 hold K {0..15},
            // lanes 16-31 hold K {16..31}.
            const __bf16* ar = sA + ((size_t)(rowA0 + (mt << 4)) << 5);
            v16bf af;
            ((int4*)&af)[0] = *(const int4*)(ar + halfB);
            ((int4*)&af)[1] = *(const int4*)(ar + halfB + 8);
            acc[mt][0] = __builtin_amdgcn_wmma_f32_16x16x32_bf16(
                false, af, false, bf0, (short)0, acc[mt][0], false, false);
            acc[mt][1] = __builtin_amdgcn_wmma_f32_16x16x32_bf16(
                false, af, false, bf1, (short)0, acc[mt][1], false, false);
        }
    }

    // Epilogue: raw f32 (B,TOUT,256) store + per-channel sum/sumsq atomics.
    float* __restrict__ rawB = raw + (size_t)b * TOUT * kCout;
    const int rbase = (lane >> 4) << 3;            // D: hi lanes hold M = r+8
    float ps[2] = {0.f, 0.f}, pq[2] = {0.f, 0.f};
#pragma unroll
    for (int mt = 0; mt < 4; ++mt) {
#pragma unroll
        for (int n = 0; n < 2; ++n) {
            const int cc = co0 + (n << 4) + (lane & 15);
#pragma unroll
            for (int r = 0; r < 8; ++r) {
                const int t = t0 + (tw << 6) + (mt << 4) + rbase + r;
                const float v = acc[mt][n][r];
                if (t < TOUT) {
                    rawB[(size_t)t * kCout + cc] = v;
                    ps[n] += v;
                    pq[n] += v * v;
                }
            }
        }
    }
#pragma unroll
    for (int n = 0; n < 2; ++n) {
        ps[n] += __shfl_xor(ps[n], 16, 32);        // fold lane halves (wave32)
        pq[n] += __shfl_xor(pq[n], 16, 32);
    }
    if (lane < 16) {
#pragma unroll
        for (int n = 0; n < 2; ++n) {
            const int cc = co0 + (n << 4) + lane;
            atomicAdd(&stats[cc], ps[n]);
            atomicAdd(&stats[kCout + cc], pq[n]);
        }
    }
}

// ------------------------------ BN finalize --------------------------------
__global__ void enc_bn_finalize(const float* __restrict__ stats,
                                const float* __restrict__ g,
                                const float* __restrict__ be,
                                float* __restrict__ ss, float invN)
{
    const int c = threadIdx.x;                     // 256 threads, 1 block
    const float mu  = stats[c] * invN;
    const float var = stats[kCout + c] * invN - mu * mu;
    const float sc  = g[c] * rsqrtf(var + 1e-5f);
    ss[c]         = sc;
    ss[kCout + c] = be[c] - mu * sc;
}

// ------------------------ BN apply -> bf16 (next in) -----------------------
__global__ void enc_apply_bf16(const float4* __restrict__ raw4,
                               const float* __restrict__ ss,
                               __bf16* __restrict__ outp, size_t n4)
{
    const size_t i = blockIdx.x * (size_t)blockDim.x + threadIdx.x;
    if (i >= n4) return;
    const float4 v = raw4[i];
    const int c = (int)((i * 4) & (kCout - 1));
    float r[4] = {v.x, v.y, v.z, v.w};
    v4bf o;
#pragma unroll
    for (int j = 0; j < 4; ++j) {
        float u = r[j] * ss[c + j] + ss[kCout + c + j];
        u = (u >= 0.f) ? u : 0.9f * u;
        o[j] = (__bf16)u;
    }
    *(v4bf*)(outp + i * 4) = o;
}

// ---------------- BN apply + transpose to (B, 256, Tout) f32 ---------------
__global__ __launch_bounds__(256)
void enc_apply_final(const float* __restrict__ raw, const float* __restrict__ ss,
                     float* __restrict__ outp, int Tout)
{
    __shared__ float tile[32][33];
    const int b  = blockIdx.z;
    const int t0 = blockIdx.x << 5;
    const int c0 = blockIdx.y << 5;
    const int tx = threadIdx.x & 31;
    const int ty = threadIdx.x >> 5;
    const float* __restrict__ rawB = raw + (size_t)b * Tout * kCout;
#pragma unroll
    for (int j = 0; j < 4; ++j) {
        const int t = t0 + ty + (j << 3);
        const int c = c0 + tx;
        float v = 0.f;
        if (t < Tout) {
            v = rawB[(size_t)t * kCout + c];
            v = v * ss[c] + ss[kCout + c];
            v = (v >= 0.f) ? v : 0.9f * v;
        }
        tile[ty + (j << 3)][tx] = v;
    }
    __syncthreads();
    float* __restrict__ outB = outp + (size_t)b * kCout * Tout;
#pragma unroll
    for (int j = 0; j < 4; ++j) {
        const int c = c0 + ty + (j << 3);
        const int t = t0 + tx;
        if (t < Tout) outB[(size_t)c * Tout + t] = tile[tx][ty + (j << 3)];
    }
}

// --------------------------------- driver ----------------------------------
extern "C" void kernel_launch(void* const* d_in, const int* in_sizes, int n_in,
                              void* d_out, int out_size, void* d_ws, size_t ws_size,
                              hipStream_t stream)
{
    (void)in_sizes; (void)n_in; (void)out_size; (void)ws_size;

    const float* X = (const float*)d_in[0];
    const float* Y = (const float*)d_in[1];

    size_t off = 0;
    auto carve = [&](size_t bytes) -> void* {
        void* p = (char*)d_ws + off;
        off += (bytes + 255) & ~(size_t)255;
        return p;
    };
    const size_t ELT = (size_t)kB * kT * kCout;
    __bf16* bufA  = (__bf16*)carve(ELT * 2);
    __bf16* bufB  = (__bf16*)carve(ELT * 2);
    float*  raw   = (float*) carve(ELT * 4);
    __bf16* wpk   = (__bf16*)carve((size_t)kCout * 768 * 2);
    float*  stats = (float*) carve(2 * kCout * 4);
    float*  ss    = (float*) carve(2 * kCout * 4);

    // concat -> (B, T, 64) bf16
    {
        dim3 grid(kT / 32, 2, kB);
        enc_concat_bf16<<<grid, 256, 0, stream>>>(X, Y, bufA);
    }

    struct Stage { int Cin, dil, Tin, Tout; };
    const Stage st[4] = {
        { 64, 1, 4096, 4094},
        {256, 2, 4094, 4090},
        {256, 4, 4090, 4082},
        {256, 8, 4082, 4066},
    };

    __bf16* cur = bufA;
    __bf16* nxt = bufB;

    for (int i = 0; i < 4; ++i) {
        const float* W  = (const float*)d_in[2 + i * 4];
        const float* g  = (const float*)d_in[4 + i * 4];
        const float* be = (const float*)d_in[5 + i * 4];
        const int Cin = st[i].Cin;
        const int Tout = st[i].Tout;

        {   // pack weights
            const int tot = kCout * Cin * 3;
            enc_pack_w<<<(tot + 255) / 256, 256, 0, stream>>>(W, wpk, Cin);
        }
        enc_zero_f32<<<2, 256, 0, stream>>>(stats, 2 * kCout);

        {   // conv + stats (templated on stage constants)
            dim3 grid((Tout + 127) / 128, 2, kB);
            switch (i) {
            case 0: enc_conv_wmma_t< 64, 1, 4096, 4094>
                        <<<grid, 256, 0, stream>>>(cur, wpk, raw, stats); break;
            case 1: enc_conv_wmma_t<256, 2, 4094, 4090>
                        <<<grid, 256, 0, stream>>>(cur, wpk, raw, stats); break;
            case 2: enc_conv_wmma_t<256, 4, 4090, 4082>
                        <<<grid, 256, 0, stream>>>(cur, wpk, raw, stats); break;
            default: enc_conv_wmma_t<256, 8, 4082, 4066>
                        <<<grid, 256, 0, stream>>>(cur, wpk, raw, stats); break;
            }
        }
        enc_bn_finalize<<<1, kCout, 0, stream>>>(stats, g, be, ss,
                                                 1.0f / (float)(kB * Tout));

        if (i < 3) {
            const size_t n4 = (size_t)kB * Tout * kCout / 4;
            enc_apply_bf16<<<(unsigned)((n4 + 255) / 256), 256, 0, stream>>>(
                (const float4*)raw, ss, nxt, n4);
            __bf16* tmp = cur; cur = nxt; nxt = tmp;
        } else {
            dim3 grid((Tout + 31) / 32, kCout / 32, kB);
            enc_apply_final<<<grid, 256, 0, stream>>>(raw, ss, (float*)d_out, Tout);
        }
    }
}
